// BranchKNNSoftmax_58617713655835
// MI455X (gfx1250) — matrix operand order repaired and checked by reading, not compiled, not run on hardware
//
#include <hip/hip_runtime.h>
#include <hip/hip_bf16.h>

// ---------------- problem constants (match reference) ----------------
#define NROWS   8192
#define DTOT    512
#define DK      128      // per-branch dim
#define NB      4        // branches
#define KNN     100      // neighbors kept
#define ALPHA   40.0f
#define ROWS_PER_BLOCK 64
#define WAVES   4        // 4 waves x 16 rows = 64 rows per block
#define SPITCH  20       // stage row pitch (floats): 16B-aligned rows, no bank conflicts

typedef __attribute__((ext_vector_type(16))) _Float16 v16h;
typedef __attribute__((ext_vector_type(8)))  _Float16 v8h;
typedef __attribute__((ext_vector_type(8)))  float    v8f;
typedef __attribute__((ext_vector_type(4)))  float    v4f;

union V16 { v16h v; v8h h[2]; };
union V4  { v4f  v; float f[4]; };

#if __has_builtin(__builtin_amdgcn_sched_barrier)
#define SCHED_BARRIER() __builtin_amdgcn_sched_barrier(0)
#else
#define SCHED_BARRIER() asm volatile("" ::: "memory")
#endif

// Async memory->LDS copy of 16 bytes (8 halfs), tracked with ASYNCcnt.
// VDST = LDS byte address (low 32 bits of generic shared pointer),
// VADDR = 64-bit global address (GV mode, saddr=off).
__device__ __forceinline__ void async_copy_b128(const _Float16* src, _Float16* dst) {
  unsigned int lds_off = (unsigned int)(unsigned long long)dst;
  unsigned long long ga = (unsigned long long)src;
  asm volatile("global_load_async_to_lds_b128 %0, %1, off"
               :: "v"(lds_off), "v"(ga) : "memory");
}

// ---------------------------------------------------------------------
// Kernel 1: gather branch columns + L2 normalize rows -> f16 Xh[b][n][128]
// ---------------------------------------------------------------------
__global__ __launch_bounds__(128)
void gather_norm(const float* __restrict__ inputs,
                 const int*   __restrict__ bidx,
                 _Float16*    __restrict__ Xh) {
  int n = blockIdx.x;        // row
  int b = blockIdx.y;        // branch
  int t = threadIdx.x;       // 0..127 = feature
  __shared__ float red[128];

  int col = bidx[b * DK + t];
  float v = inputs[(size_t)n * DTOT + col];
  red[t] = v * v;
  __syncthreads();
  for (int s = 64; s > 0; s >>= 1) {
    if (t < s) red[t] += red[t + s];
    __syncthreads();
  }
  float rn = rsqrtf(red[0] + 1e-30f);
  Xh[((size_t)b * NROWS + n) * DK + t] = (_Float16)(v * rn);
}

// ---------------------------------------------------------------------
// Kernel 2: fused WMMA similarity + running top-100 per row
//   grid  = NB * (NROWS / ROWS_PER_BLOCK) blocks
//   block = 128 threads = 4 waves; wave w owns 16 rows
//   column panels double-buffered in LDS, filled by async memory->LDS DMA
// ---------------------------------------------------------------------
__global__ __launch_bounds__(128)
void simtopk(const _Float16* __restrict__ Xh,
             float* __restrict__ topv_g,
             int*   __restrict__ topi_g) {
  __shared__ _Float16 panel[2][16 * DK];             // 2 x 4 KB, double buffer
  __shared__ float    stage[WAVES][16][SPITCH];      // 16x16 sim tile per wave
  __shared__ float    topv[ROWS_PER_BLOCK][KNN];     // 25.6 KB
  __shared__ int      topi[ROWS_PER_BLOCK][KNN];     // 25.6 KB
  __shared__ float    rowmin[ROWS_PER_BLOCK];
  __shared__ int      rowminpos[ROWS_PER_BLOCK];
  __shared__ int      rowcnt[ROWS_PER_BLOCK];

  const int blocksPerBranch = NROWS / ROWS_PER_BLOCK;
  int b  = blockIdx.x / blocksPerBranch;
  int rp = blockIdx.x % blocksPerBranch;
  int rowBase = rp * ROWS_PER_BLOCK;
  const _Float16* Xb = Xh + (size_t)b * NROWS * DK;

  int t    = threadIdx.x;
  int wave = t >> 5;
  int lane = t & 31;

  // cooperative panel staging pattern: thread t owns column (t>>3),
  // halfs [(t&7)*16, (t&7)*16+15] of that column's 128-dim K vector.
  int pcol = t >> 3;
  int ph0  = (t & 7) * 16;

  for (int r = t; r < ROWS_PER_BLOCK; r += 128) {
    rowcnt[r] = 0; rowmin[r] = -1e30f; rowminpos[r] = 0;
  }

  // ---- prologue: async-stage column tile 0 into buffer 0
  {
    const _Float16* src = Xb + (size_t)pcol * DK + ph0;
    _Float16* dst = &panel[0][pcol * DK + ph0];
    async_copy_b128(src, dst);
    async_copy_b128(src + 8, dst + 8);
  }

  // ---- load this wave's A fragments once (16 rows x K=128, 4 chunks of 32)
  // CDNA5 16-bit A layout: lane l holds row M=l%16; lanes 0-15 get halves
  // {K..K+7, K+16..K+23}, lanes 16-31 get {K+8..K+15, K+24..K+31}.
  int arow = rowBase + wave * 16 + (lane & 15);
  const _Float16* Arow = Xb + (size_t)arow * DK;
  int aShift = (lane < 16) ? 0 : 8;
  V16 afrag[4];
#pragma unroll
  for (int c = 0; c < 4; ++c) {
    afrag[c].h[0] = *(const v8h*)(Arow + c * 32 + aShift);
    afrag[c].h[1] = *(const v8h*)(Arow + c * 32 + aShift + 16);
  }

  const int NT = NROWS / 16;              // 512 column tiles
  for (int ct = 0; ct < NT; ++ct) {
    // issue async stage of NEXT tile into the other buffer, then wait for
    // the CURRENT tile (the 2 newest async ops are still allowed in flight)
    if (ct + 1 < NT) {
      const _Float16* src = Xb + (size_t)((ct + 1) * 16 + pcol) * DK + ph0;
      _Float16* dst = &panel[(ct + 1) & 1][pcol * DK + ph0];
      async_copy_b128(src, dst);
      async_copy_b128(src + 8, dst + 8);
      __builtin_prefetch(Xb + (size_t)((ct + 2) * 16 + pcol) * DK, 0, 0);
      asm volatile("s_wait_asynccnt 0x2" ::: "memory");
    } else {
      asm volatile("s_wait_asynccnt 0x0" ::: "memory");
    }
    __syncthreads();                      // whole current panel visible

    const _Float16* cur = &panel[ct & 1][0];
    int colBase = ct * 16;

    // ---- load ALL 4 B-chunks (8x ds_load_b128) before any WMMA issues;
    // sched_barrier keeps the backend from re-interleaving load->wait->wmma.
    // B(k,n)=X[colBase+n][k]; lanes 0-15 hold K={0..7,16..23} of col n, 16-31 K={8..15,24..31}.
    int bn  = lane & 15;
    int bks = (lane < 16) ? 0 : 8;
    V16 bf[4];
#pragma unroll
    for (int c = 0; c < 4; ++c) {
      const _Float16* bp = cur + bn * DK + c * 32 + bks;
      bf[c].h[0] = *(const v8h*)(bp);
      bf[c].h[1] = *(const v8h*)(bp + 16);
    }
    SCHED_BARRIER();
    v8f acc = {};
#pragma unroll
    for (int c = 0; c < 4; ++c) {
      acc = __builtin_amdgcn_wmma_f32_16x16x32_f16(
          /*neg_a=*/false, afrag[c].v, /*neg_b=*/false, bf[c].v,
          /*c_mod=*/(short)0, acc, /*reuse_a=*/false, /*reuse_b=*/false);
    }
    SCHED_BARRIER();

    // spill C tile: VGPR v, lanes 0-15 -> M=v, lanes 16-31 -> M=v+8; N=lane%16
#pragma unroll
    for (int v = 0; v < 8; ++v) {
      int M = v + ((lane >> 4) << 3);
      stage[wave][M][lane & 15] = acc[v];
    }
    __syncthreads();

    // ---- top-100 maintenance: lane m (0..15) owns row M=m of this wave.
    // Pull the 16 candidates with 4x ds_load_b128 into registers first.
    if (lane < 16) {
      int r    = wave * 16 + lane;        // row within block
      int grow = rowBase + r;             // global row (diag mask)
      const float* srow = &stage[wave][lane][0];
      V4 sv[4];
#pragma unroll
      for (int q = 0; q < 4; ++q) sv[q].v = *(const v4f*)(srow + q * 4);

      float rmin = rowmin[r];
      int   rmp  = rowminpos[r];
      int   cnt  = rowcnt[r];
#pragma unroll
      for (int j = 0; j < 16; ++j) {
        int col = colBase + j;
        float val = sv[j >> 2].f[j & 3];
        if (col == grow) continue;        // self-similarity masked
        if (cnt < KNN) {
          topv[r][cnt] = val; topi[r][cnt] = col;
          if (cnt == 0 || val < rmin) { rmin = val; rmp = cnt; }
          ++cnt;
        } else if (val > rmin) {
          topv[r][rmp] = val; topi[r][rmp] = col;
          rmin = topv[r][0]; rmp = 0;
#pragma unroll 1
          for (int q = 1; q < KNN; ++q) {
            float tv = topv[r][q];
            if (tv < rmin) { rmin = tv; rmp = q; }
          }
        }
      }
      rowmin[r] = rmin; rowminpos[r] = rmp; rowcnt[r] = cnt;
    }
    __syncthreads();                      // protects panel/stage reuse
  }

  // ---- write per-row top-100 to workspace
  for (int r = 0; r < ROWS_PER_BLOCK; ++r) {
    size_t base = ((size_t)b * NROWS + (rowBase + r)) * KNN;
    for (int q = t; q < KNN; q += 128) {
      topv_g[base + q] = topv[r][q];
      topi_g[base + q] = topi[r][q];
    }
  }
}

// ---------------------------------------------------------------------
// Kernel 3: per-row KNN-softmax loss + stats, atomically accumulated
// accum[b*8 + {0:lossSum,1:hasPosCnt,2:posSum,3:posCnt,4:negSum,5:top1Hit}]
// ---------------------------------------------------------------------
__global__ __launch_bounds__(128)
void finalize(const float* __restrict__ topv_g,
              const int*   __restrict__ topi_g,
              const int*   __restrict__ targets,
              float* __restrict__ accum) {
  int gid = blockIdx.x * blockDim.x + threadIdx.x;
  if (gid >= NB * NROWS) return;
  int b   = gid / NROWS;
  int row = gid % NROWS;
  const float* tv = topv_g + (size_t)gid * KNN;
  const int*   ti = topi_g + (size_t)gid * KNN;
  int mylab = targets[row];

  float sumall = 0.f, sumpos = 0.f;
  int   cntpos = 0, argmax = 0;
  float maxv = -1e30f, mAll = -1e30f, mPos = -1e30f;
  for (int q = 0; q < KNN; ++q) {
    float v = tv[q];
    sumall += v;
    if (v > maxv) { maxv = v; argmax = q; }
    float l = ALPHA * v;
    if (l > mAll) mAll = l;
    if (targets[ti[q]] == mylab) {
      sumpos += v; ++cntpos;
      if (l > mPos) mPos = l;
    }
  }
  float den = 0.f, num = 0.f;
  for (int q = 0; q < KNN; ++q) {
    float l = ALPHA * tv[q];
    den += __expf(l - mAll);
    if (targets[ti[q]] == mylab) num += __expf(l - mPos);
  }
  bool haspos = cntpos > 0;
  float loss_i = haspos ? ((mAll + logf(den)) - (mPos + logf(num))) : 0.f;
  float top1 = (targets[ti[argmax]] == mylab) ? 1.f : 0.f;

  atomicAdd(&accum[b * 8 + 0], loss_i);
  atomicAdd(&accum[b * 8 + 1], haspos ? 1.f : 0.f);
  atomicAdd(&accum[b * 8 + 2], sumpos);
  atomicAdd(&accum[b * 8 + 3], (float)cntpos);
  atomicAdd(&accum[b * 8 + 4], sumall - sumpos);
  atomicAdd(&accum[b * 8 + 5], top1);
}

// ---------------------------------------------------------------------
// Kernel 4: combine per-branch accumulators into the 4 output scalars
// ---------------------------------------------------------------------
__global__ void reduce_out(const float* __restrict__ accum,
                           float* __restrict__ out) {
  if (threadIdx.x != 0 || blockIdx.x != 0) return;
  float loss = 0.f, accv = 0.f, posd = 0.f, negd = 0.f;
  for (int b = 0; b < NB; ++b) {
    float lsum = accum[b * 8 + 0];
    float hcnt = accum[b * 8 + 1];
    float psum = accum[b * 8 + 2];
    float pcnt = accum[b * 8 + 3];
    float nsum = accum[b * 8 + 4];
    float a1   = accum[b * 8 + 5];
    loss += lsum / fmaxf(hcnt, 1.f);
    accv += a1;
    posd += psum / fmaxf(pcnt, 1.f);
    float ncnt = (float)NROWS * (float)KNN - pcnt;
    negd += nsum / fmaxf(ncnt, 1.f);
  }
  out[0] = loss * 0.25f;
  out[1] = accv / (float)(NB * NROWS);
  out[2] = posd * 0.25f;
  out[3] = negd * 0.25f;
}

__global__ void initacc(float* __restrict__ accum) {
  if (threadIdx.x < NB * 8) accum[threadIdx.x] = 0.f;
}

// ---------------------------------------------------------------------
extern "C" void kernel_launch(void* const* d_in, const int* in_sizes, int n_in,
                              void* d_out, int out_size, void* d_ws, size_t ws_size,
                              hipStream_t stream) {
  const float* inputs  = (const float*)d_in[0];   // [8192, 512] f32
  const int*   targets = (const int*)  d_in[1];   // [8192] i32
  const int*   bidx    = (const int*)  d_in[2];   // [4, 128] i32
  float* out = (float*)d_out;                     // 4 scalars

  // workspace layout (all 256B aligned)
  char* ws = (char*)d_ws;
  _Float16* Xh   = (_Float16*)ws;                           // 8 MB
  size_t off = (size_t)NB * NROWS * DK * sizeof(_Float16);
  float* topv_g  = (float*)(ws + off);                      // 13.1 MB
  off += (size_t)NB * NROWS * KNN * sizeof(float);
  int*   topi_g  = (int*)(ws + off);                        // 13.1 MB
  off += (size_t)NB * NROWS * KNN * sizeof(int);
  float* accum   = (float*)(ws + off);                      // 32 floats

  initacc<<<1, 64, 0, stream>>>(accum);
  gather_norm<<<dim3(NROWS, NB), 128, 0, stream>>>(inputs, bidx, Xh);
  simtopk<<<NB * (NROWS / ROWS_PER_BLOCK), 128, 0, stream>>>(Xh, topv_g, topi_g);
  finalize<<<(NB * NROWS + 127) / 128, 128, 0, stream>>>(topv_g, topi_g, targets, accum);
  reduce_out<<<1, 1, 0, stream>>>(accum, out);
}
